// ReparamGaussianMoE_48086453846278
// MI455X (gfx1250) — compile-verified, matrix operand order checked by reading, not run on hardware
//
#include <hip/hip_runtime.h>

// ReparamGaussianMoE on MI455X (gfx1250, wave32, WMMA bf16 16x16x32).
// D=1024, H=2048, E=8, N=4096, TAU=0.1. Compute-bound (~570 GFLOP vs ~420MB).

typedef __attribute__((ext_vector_type(16))) __bf16 v16bf;
typedef __attribute__((ext_vector_type(8)))  float  v8f;
typedef __attribute__((ext_vector_type(4)))  int    v4i;

#define D_DIM 1024
#define H_DIM 2048
#define E_DIM 8
#define N_TOK 4096
#define TAU_F 0.1f
#define LPAD  40   // padded LDS row (bf16 elems): 80B stride, 16B aligned, conflict-free

union FragB16 { uint4 q[2]; v16bf v; };
union PackB16 { __bf16 h[16]; uint4 q[2]; };

// ---- optional async global->LDS path (ASYNCcnt) ----
#if __has_builtin(__builtin_amdgcn_global_load_async_to_lds_b128) && \
    __has_builtin(__builtin_amdgcn_s_wait_asynccnt)
#define HAVE_ASYNC_LDS 1
__device__ __forceinline__ void async_cp16(const void* g, void* l) {
    __builtin_amdgcn_global_load_async_to_lds_b128((v4i*)g, (v4i*)l, 0, 0);
}
__device__ __forceinline__ void async_wait0() { __builtin_amdgcn_s_wait_asynccnt(0); }
#else
__device__ __forceinline__ void async_cp16(const void* g, void* l) {
    *(uint4*)l = *(const uint4*)g;
}
__device__ __forceinline__ void async_wait0() {}
#endif

// ---------------- kernel 1: fp32 -> bf16 convert (x) ----------------
__global__ __launch_bounds__(256) void k_cvt_bf16(const float* __restrict__ src,
                                                  __bf16* __restrict__ dst, int n) {
    int i = (blockIdx.x * 256 + threadIdx.x) * 4;
    if (i + 3 < n) {
        float4 f = *(const float4*)(src + i);
        dst[i + 0] = (__bf16)f.x;
        dst[i + 1] = (__bf16)f.y;
        dst[i + 2] = (__bf16)f.z;
        dst[i + 3] = (__bf16)f.w;
    }
}

// ---------------- kernel 2: C = relu(A_bf16 @ W_f32 + bias) -> bf16 ----------------
// Tile: 128x128, BK=32. 256 threads = 8 waves, wave grid 4x2, each wave 2x4 16x16 tiles.
__global__ __launch_bounds__(256)
void k_gemm_relu(const __bf16* __restrict__ A, const float* __restrict__ W,
                 const float* __restrict__ bias, __bf16* __restrict__ Cout,
                 int K, int Nn) {
    __shared__ __bf16 As[128][LPAD];   // activations [m][k]
    __shared__ __bf16 Bs[128][LPAD];   // weights transposed [n][k]

    const int tid  = threadIdx.x;
    const int lane = tid & 31, wave = tid >> 5;
    const int wm   = wave & 3, wn = wave >> 2;          // 4x2 wave grid
    const int ntiles = Nn >> 7;
    const int m0 = (blockIdx.x / ntiles) << 7;
    const int nb = (blockIdx.x % ntiles) << 7;

    v8f acc[2][4] = {};

    const int ar = tid >> 1, ah = (tid & 1) * 16;       // A loader: row, 16-elem K seg
    const int bn = tid & 127, bkh = (tid >> 7) * 16;    // B loader: n-row, 16-elem K seg
    const int lm = lane & 15, kg = (lane >> 4) * 8;     // fragment addressing

    for (int kb = 0; kb < K; kb += 32) {
        // A tile: 32B per thread, bf16 row-major (async DMA to LDS when available)
        const __bf16* asrc = A + (size_t)(m0 + ar) * K + kb + ah;
        async_cp16(asrc,     &As[ar][ah]);
        async_cp16(asrc + 8, &As[ar][ah + 8]);

        // W tile: fp32 [k][n] -> bf16 LDS [n][k]; coalesced reads (n across lanes),
        // packed b128 LDS stores into padded rows (no bank conflicts)
        if (kb + 32 < K)
            __builtin_prefetch(W + (size_t)(kb + 32 + bkh) * Nn + nb + bn, 0, 3);
        PackB16 pw;
#pragma unroll
        for (int kk = 0; kk < 16; ++kk)
            pw.h[kk] = (__bf16)W[(size_t)(kb + bkh + kk) * Nn + nb + bn];
        *(uint4*)&Bs[bn][bkh]     = pw.q[0];
        *(uint4*)&Bs[bn][bkh + 8] = pw.q[1];

        async_wait0();
        __syncthreads();

        // Fragments per ISA 16-bit layout: halves 0..7 = K[kg,kg+8), 8..15 = K[kg+16,kg+24)
        FragB16 a[2], b[4];
#pragma unroll
        for (int i = 0; i < 2; ++i) {
            const __bf16* p = &As[wm * 32 + i * 16 + lm][0];
            a[i].q[0] = *(const uint4*)(p + kg);
            a[i].q[1] = *(const uint4*)(p + kg + 16);
        }
#pragma unroll
        for (int j = 0; j < 4; ++j) {
            const __bf16* p = &Bs[wn * 64 + j * 16 + lm][0];
            b[j].q[0] = *(const uint4*)(p + kg);
            b[j].q[1] = *(const uint4*)(p + kg + 16);
        }
#pragma unroll
        for (int i = 0; i < 2; ++i)
#pragma unroll
            for (int j = 0; j < 4; ++j)
                acc[i][j] = __builtin_amdgcn_wmma_f32_16x16x32_bf16(
                    false, a[i].v, false, b[j].v, (short)0, acc[i][j], false, false);
        __syncthreads();
    }

    // Epilogue: bias + ReLU, store bf16. C layout: row = r + 8*(lane>>4), col = lane&15.
    const int lr = (lane >> 4) * 8;
#pragma unroll
    for (int i = 0; i < 2; ++i)
#pragma unroll
        for (int j = 0; j < 4; ++j) {
            const int col = nb + wn * 64 + j * 16 + lm;
            const float bvl = bias[col];
#pragma unroll
            for (int r = 0; r < 8; ++r) {
                const int row = m0 + wm * 32 + i * 16 + lr + r;
                Cout[(size_t)row * Nn + col] = (__bf16)fmaxf(acc[i][j][r] + bvl, 0.0f);
            }
        }
}

// ---------------- kernel 3: gate layer2 + gumbel softmax ----------------
__global__ __launch_bounds__(256)
void k_gate(const __bf16* __restrict__ hg, const float* __restrict__ Wg2,
            const float* __restrict__ bg2, const float* __restrict__ gu,
            float* __restrict__ wout) {
    const int lane = threadIdx.x & 31;
    const int wave = threadIdx.x >> 5;
    const int token = blockIdx.x * 8 + wave;

    float acc[E_DIM] = {};
    for (int h = lane; h < H_DIM; h += 32) {
        const float hv = (float)hg[(size_t)token * H_DIM + h];
        const float4* wrow = (const float4*)(Wg2 + (size_t)h * E_DIM);
        const float4 w0 = wrow[0], w1 = wrow[1];
        acc[0] += hv * w0.x; acc[1] += hv * w0.y;
        acc[2] += hv * w0.z; acc[3] += hv * w0.w;
        acc[4] += hv * w1.x; acc[5] += hv * w1.y;
        acc[6] += hv * w1.z; acc[7] += hv * w1.w;
    }
#pragma unroll
    for (int e = 0; e < E_DIM; ++e)
#pragma unroll
        for (int off = 16; off > 0; off >>= 1)
            acc[e] += __shfl_xor(acc[e], off, 32);

    if (lane == 0) {
        float z[E_DIM], zmax = -3.0e38f;
#pragma unroll
        for (int e = 0; e < E_DIM; ++e) {
            const float g = -__logf(-__logf(gu[(size_t)token * E_DIM + e]));
            z[e] = (acc[e] + bg2[e] + g) * (1.0f / TAU_F);
            zmax = fmaxf(zmax, z[e]);
        }
        float se = 0.0f;
#pragma unroll
        for (int e = 0; e < E_DIM; ++e) { z[e] = __expf(z[e] - zmax); se += z[e]; }
        const float inv = 1.0f / se;
#pragma unroll
        for (int e = 0; e < E_DIM; ++e)
            wout[(size_t)token * E_DIM + e] = z[e] * inv;
    }
}

// ---------------- kernel 4: dual layer-2 GEMM + reparam + gating ----------------
// mu = hm @ Wm2 + bm2 ; lv = hv @ Wv2 + bv2 ; out (+)= w[n,e]*(mu + eps*exp(0.5*lv))
__global__ __launch_bounds__(256)
void k_gemm2_sample(const __bf16* __restrict__ Am, const __bf16* __restrict__ Av,
                    const float* __restrict__ Wm, const float* __restrict__ Wv,
                    const float* __restrict__ bm, const float* __restrict__ bv,
                    const float* __restrict__ eps, const float* __restrict__ gate_w,
                    float* __restrict__ out, int expert, int accumulate) {
    const int K = H_DIM, Nn = D_DIM;
    __shared__ __bf16 Ams[128][LPAD], Avs[128][LPAD];
    __shared__ __bf16 Bms[128][LPAD], Bvs[128][LPAD];

    const int tid  = threadIdx.x;
    const int lane = tid & 31, wave = tid >> 5;
    const int wm   = wave & 3, wn = wave >> 2;
    const int ntiles = Nn >> 7;
    const int m0 = (blockIdx.x / ntiles) << 7;
    const int nb = (blockIdx.x % ntiles) << 7;

    v8f accm[2][4] = {}, accv[2][4] = {};

    const int ar = tid >> 1, ah = (tid & 1) * 16;
    const int bn = tid & 127, bkh = (tid >> 7) * 16;
    const int lm = lane & 15, kg = (lane >> 4) * 8;

    for (int kb = 0; kb < K; kb += 32) {
        const __bf16* s0 = Am + (size_t)(m0 + ar) * K + kb + ah;
        const __bf16* s1 = Av + (size_t)(m0 + ar) * K + kb + ah;
        async_cp16(s0,     &Ams[ar][ah]);
        async_cp16(s0 + 8, &Ams[ar][ah + 8]);
        async_cp16(s1,     &Avs[ar][ah]);
        async_cp16(s1 + 8, &Avs[ar][ah + 8]);

        if (kb + 32 < K) {
            __builtin_prefetch(Wm + (size_t)(kb + 32 + bkh) * Nn + nb + bn, 0, 3);
            __builtin_prefetch(Wv + (size_t)(kb + 32 + bkh) * Nn + nb + bn, 0, 3);
        }
        PackB16 pm, pv;
#pragma unroll
        for (int kk = 0; kk < 16; ++kk) {
            const size_t wi = (size_t)(kb + bkh + kk) * Nn + nb + bn;
            pm.h[kk] = (__bf16)Wm[wi];
            pv.h[kk] = (__bf16)Wv[wi];
        }
        *(uint4*)&Bms[bn][bkh]     = pm.q[0];
        *(uint4*)&Bms[bn][bkh + 8] = pm.q[1];
        *(uint4*)&Bvs[bn][bkh]     = pv.q[0];
        *(uint4*)&Bvs[bn][bkh + 8] = pv.q[1];

        async_wait0();
        __syncthreads();

        FragB16 am[2], av[2], bmf[4], bvf[4];
#pragma unroll
        for (int i = 0; i < 2; ++i) {
            const __bf16* p0 = &Ams[wm * 32 + i * 16 + lm][0];
            const __bf16* p1 = &Avs[wm * 32 + i * 16 + lm][0];
            am[i].q[0] = *(const uint4*)(p0 + kg);
            am[i].q[1] = *(const uint4*)(p0 + kg + 16);
            av[i].q[0] = *(const uint4*)(p1 + kg);
            av[i].q[1] = *(const uint4*)(p1 + kg + 16);
        }
#pragma unroll
        for (int j = 0; j < 4; ++j) {
            const __bf16* p0 = &Bms[wn * 64 + j * 16 + lm][0];
            const __bf16* p1 = &Bvs[wn * 64 + j * 16 + lm][0];
            bmf[j].q[0] = *(const uint4*)(p0 + kg);
            bmf[j].q[1] = *(const uint4*)(p0 + kg + 16);
            bvf[j].q[0] = *(const uint4*)(p1 + kg);
            bvf[j].q[1] = *(const uint4*)(p1 + kg + 16);
        }
#pragma unroll
        for (int i = 0; i < 2; ++i)
#pragma unroll
            for (int j = 0; j < 4; ++j) {
                accm[i][j] = __builtin_amdgcn_wmma_f32_16x16x32_bf16(
                    false, am[i].v, false, bmf[j].v, (short)0, accm[i][j], false, false);
                accv[i][j] = __builtin_amdgcn_wmma_f32_16x16x32_bf16(
                    false, av[i].v, false, bvf[j].v, (short)0, accv[i][j], false, false);
            }
        __syncthreads();
    }

    // Epilogue: reparameterized sample, gate-weight, accumulate into out.
    const int lr = (lane >> 4) * 8;
#pragma unroll
    for (int i = 0; i < 2; ++i)
#pragma unroll
        for (int j = 0; j < 4; ++j) {
            const int col = nb + wn * 64 + j * 16 + lm;
            const float bmc = bm[col], bvc = bv[col];
#pragma unroll
            for (int r = 0; r < 8; ++r) {
                const int row = m0 + wm * 32 + i * 16 + lr + r;
                const float mu = accm[i][j][r] + bmc;
                const float lv = accv[i][j][r] + bvc;
                const size_t ei = ((size_t)row * E_DIM + expert) * D_DIM + col;
                const float ep = __builtin_nontemporal_load(eps + ei);  // single-use stream
                const float s  = mu + ep * __expf(0.5f * lv);
                const float gw = gate_w[(size_t)row * E_DIM + expert];
                const size_t oi = (size_t)row * D_DIM + col;
                if (accumulate) out[oi] += gw * s;
                else            out[oi]  = gw * s;
            }
        }
}

// ---------------- host launcher ----------------
extern "C" void kernel_launch(void* const* d_in, const int* in_sizes, int n_in,
                              void* d_out, int out_size, void* d_ws, size_t ws_size,
                              hipStream_t stream) {
    const float* x   = (const float*)d_in[0];
    const float* gu  = (const float*)d_in[1];
    const float* eps = (const float*)d_in[2];
    const float* Wg1 = (const float*)d_in[3];
    const float* bg1 = (const float*)d_in[4];
    const float* Wg2 = (const float*)d_in[5];
    const float* bg2 = (const float*)d_in[6];
    const float* Wm1 = (const float*)d_in[7];
    const float* bm1 = (const float*)d_in[8];
    const float* Wm2 = (const float*)d_in[9];
    const float* bm2 = (const float*)d_in[10];
    const float* Wv1 = (const float*)d_in[11];
    const float* bv1 = (const float*)d_in[12];
    const float* Wv2 = (const float*)d_in[13];
    const float* bv2 = (const float*)d_in[14];
    float* out = (float*)d_out;

    char* ws = (char*)d_ws;
    __bf16* xb = (__bf16*)ws; ws += (size_t)N_TOK * D_DIM * sizeof(__bf16);
    __bf16* hg = (__bf16*)ws; ws += (size_t)N_TOK * H_DIM * sizeof(__bf16);
    __bf16* hm = (__bf16*)ws; ws += (size_t)N_TOK * H_DIM * sizeof(__bf16);
    __bf16* hv = (__bf16*)ws; ws += (size_t)N_TOK * H_DIM * sizeof(__bf16);
    float*  gw = (float*)ws;  // N*E

    // 1) x -> bf16
    k_cvt_bf16<<<(N_TOK * D_DIM) / (4 * 256), 256, 0, stream>>>(x, xb, N_TOK * D_DIM);

    const int g1 = (N_TOK / 128) * (H_DIM / 128);  // layer-1 grids (K=D, N=H)
    const int g2 = (N_TOK / 128) * (D_DIM / 128);  // layer-2 grids (K=H, N=D)

    // 2) gate hidden + gumbel softmax weights
    k_gemm_relu<<<g1, 256, 0, stream>>>(xb, Wg1, bg1, hg, D_DIM, H_DIM);
    k_gate<<<N_TOK / 8, 256, 0, stream>>>(hg, Wg2, bg2, gu, gw);

    // 3) experts: layer1 (mu & var), then fused layer2 + reparam + gated accumulate
    for (int e = 0; e < E_DIM; ++e) {
        k_gemm_relu<<<g1, 256, 0, stream>>>(xb, Wm1 + (size_t)e * D_DIM * H_DIM,
                                            bm1 + (size_t)e * H_DIM, hm, D_DIM, H_DIM);
        k_gemm_relu<<<g1, 256, 0, stream>>>(xb, Wv1 + (size_t)e * D_DIM * H_DIM,
                                            bv1 + (size_t)e * H_DIM, hv, D_DIM, H_DIM);
        k_gemm2_sample<<<g2, 256, 0, stream>>>(hm, hv,
                                               Wm2 + (size_t)e * H_DIM * D_DIM,
                                               Wv2 + (size_t)e * H_DIM * D_DIM,
                                               bm2 + (size_t)e * D_DIM,
                                               bv2 + (size_t)e * D_DIM,
                                               eps, gw, out, e, e != 0);
    }
}
